// MultiHeadAttention_17901423690038
// MI455X (gfx1250) — matrix operand order repaired and checked by reading, not compiled, not run on hardware
//
#include <hip/hip_runtime.h>

// ---------------------------------------------------------------------------
// Fused additive multi-head attention for MI455X (gfx1250, wave32, WMMA).
//   k = key @ Wk^T + bk        (137 GFLOP -> bf16 WMMA 16x16x32)
//   add = sum_d nv[d]*tanh(k+q); softmax over heads; ctx = w @ key
// Fully fused per 64-row key tile: key read from HBM once.
// Staging uses CDNA5 GLOBAL_LOAD_ASYNC_TO_LDS (ASYNCcnt) into an LDS bounce
// buffer, then LDS->LDS fp32->bf16 conversion into the padded WMMA tile.
// ---------------------------------------------------------------------------

typedef __attribute__((ext_vector_type(16))) __bf16 v16bf;
typedef __attribute__((ext_vector_type(8)))  __bf16 v8bf;
typedef __attribute__((ext_vector_type(8)))  float  v8f;

#define NB   64     // batch
#define TK   2048   // key time steps
#define KD   512    // key feature dim (== QD)
#define UD   1024   // projection units
#define HH   8      // heads
#define DH   128    // per-head dim
#define TM   64     // t-rows per workgroup
#define CROWS 16    // rows per async staging chunk
#define LDSROW 520  // padded bf16 row stride (1040 B, 16B aligned)

union AFrag { v16bf v; v8bf h[2]; };

// --------------------------- prep kernels ----------------------------------

__global__ void cvt_wk_bf16_kernel(const float* __restrict__ w, __bf16* __restrict__ o) {
  int i = blockIdx.x * blockDim.x + threadIdx.x;        // 131072 threads, 4 elems each
  float4 f = ((const float4*)w)[i];
  __bf16* d = o + 4 * (size_t)i;
  d[0] = (__bf16)f.x; d[1] = (__bf16)f.y; d[2] = (__bf16)f.z; d[3] = (__bf16)f.w;
}

__global__ void normv_kernel(const float* __restrict__ v, float* __restrict__ nv) {
  __shared__ float red[DH];
  int t = threadIdx.x;                                   // 128 threads
  float x = v[t];
  red[t] = x * x;
  __syncthreads();
  for (int s = DH / 2; s > 0; s >>= 1) {
    if (t < s) red[t] += red[t + s];
    __syncthreads();
  }
  // g * v * rsqrt(sum v^2), g = 1/sqrt(128)
  nv[t] = x * rsqrtf(red[0]) * 0.08838834764831845f;
}

__global__ __launch_bounds__(256) void qproj_kernel(
    const float* __restrict__ query, const float* __restrict__ Wq,
    const float* __restrict__ bq, float* __restrict__ qout) {
  int b = blockIdx.x, tid = threadIdx.x;
  __shared__ float sq[KD];
  for (int i = tid; i < KD; i += 256) sq[i] = query[(size_t)b * KD + i];
  __syncthreads();
  for (int u = tid; u < UD; u += 256) {
    const float4* wr = (const float4*)(Wq + (size_t)u * KD);
    const float4* qr = (const float4*)sq;
    float s = bq[u];
#pragma unroll 4
    for (int k = 0; k < KD / 4; ++k) {
      float4 w = wr[k], q4 = qr[k];
      s += w.x * q4.x + w.y * q4.y + w.z * q4.z + w.w * q4.w;
    }
    qout[(size_t)b * UD + u] = s;
  }
}

__global__ void zero_out_kernel(float* __restrict__ out) {
  int i = blockIdx.x * blockDim.x + threadIdx.x;         // 65536 threads * float4
  float4 z = {0.f, 0.f, 0.f, 0.f};
  ((float4*)out)[i] = z;
}

// --------------------------- main fused kernel -----------------------------

__global__ __launch_bounds__(256) void attn_fused_kernel(
    const float* __restrict__ key,
    const __bf16* __restrict__ wkb,     // Wk in bf16, row-major [UD][KD]
    const float* __restrict__ qbuf,     // q = query@Wq^T + bq, [NB][UD]
    const float* __restrict__ nvbuf,    // normed projection vector [DH]
    const float* __restrict__ bk,       // [UD]
    float* __restrict__ out) {          // [NB][HH*KD]
  __shared__ __attribute__((aligned(32))) __bf16 sKey[TM * LDSROW];
  __shared__ __attribute__((aligned(16))) float  sStage[CROWS * KD];  // async bounce
  __shared__ float sAdd[TM * HH];
  __shared__ float sW[TM * HH];

  const int tid  = threadIdx.x;
  const int b    = blockIdx.y;
  const int t0   = blockIdx.x * TM;
  const int lane = tid & 31;
  const int wv   = tid >> 5;            // wave id 0..7
  const int lr   = lane & 15;
  const int half = lane >> 4;
  const int nw   = wv * 16;             // this wave's head-local N chunk

  const float* keyTile = key + ((size_t)b * TK + t0) * KD;

  for (int i = tid; i < TM * HH; i += 256) sAdd[i] = 0.0f;

  // ---- stage key tile fp32 -> bf16 LDS via async-to-LDS bounce chunks ----
  for (int c = 0; c < TM / CROWS; ++c) {
    // 16 rows * 512 f32 = 2048 x 16B transfers, 8 per thread (EXEC all-ones)
    for (int i = tid; i < CROWS * (KD / 4); i += 256) {
      const float* g = keyTile + ((size_t)(c * CROWS + (i >> 7))) * KD + (i & 127) * 4;
      unsigned int ldsoff = (unsigned int)(unsigned long long)(&sStage[i * 4]);
      asm volatile("global_load_async_to_lds_b128 %0, %1, off"
                   :: "v"(ldsoff), "v"(g) : "memory");
    }
    asm volatile("s_wait_asynccnt 0x0" ::: "memory");
    __syncthreads();                    // everyone's async data visible
    for (int i = tid; i < CROWS * (KD / 4); i += 256) {
      int row = i >> 7, c4 = i & 127;   // KD/4 == 128 float4 per row
      float4 f = ((const float4*)sStage)[i];
      __bf16* dst = &sKey[(c * CROWS + row) * LDSROW + c4 * 4];
      dst[0] = (__bf16)f.x; dst[1] = (__bf16)f.y;
      dst[2] = (__bf16)f.z; dst[3] = (__bf16)f.w;
    }
    __syncthreads();                    // before bounce buffer is reused
  }

  const int   dloc = nw + lr;           // head-local dim handled by this lane
  const float nvd  = nvbuf[dloc];

#pragma unroll 1
  for (int h = 0; h < HH; ++h) {
    const int u = h * DH + dloc;
    const __bf16* wrow = wkb + (size_t)u * KD;
    if (h + 1 < HH) __builtin_prefetch(wrow + (size_t)DH * KD, 0, 1);

    v8f acc0 = {}, acc1 = {}, acc2 = {}, acc3 = {};
#pragma unroll 1
    for (int kk = 0; kk < KD; kk += 32) {
      // B fragment (32x16 bf16) per ISA layout: two 8-elem K chunks per lane
      AFrag bf;
      bf.h[0] = *(const v8bf*)(wrow + kk + half * 8);
      bf.h[1] = *(const v8bf*)(wrow + kk + 16 + half * 8);
      v16bf bfrag = bf.v;
      // A fragments (16x32 bf16) per ISA layout: two 8-elem K chunks per lane
      AFrag a;
      const __bf16* ar = &sKey[lr * LDSROW + kk + half * 8];
      a.h[0] = *(const v8bf*)ar; a.h[1] = *(const v8bf*)(ar + 16);
      acc0 = __builtin_amdgcn_wmma_f32_16x16x32_bf16(false, a.v, false, bfrag,
                                                     (short)0, acc0, false, false);
      ar += 16 * LDSROW;
      a.h[0] = *(const v8bf*)ar; a.h[1] = *(const v8bf*)(ar + 16);
      acc1 = __builtin_amdgcn_wmma_f32_16x16x32_bf16(false, a.v, false, bfrag,
                                                     (short)0, acc1, false, false);
      ar += 16 * LDSROW;
      a.h[0] = *(const v8bf*)ar; a.h[1] = *(const v8bf*)(ar + 16);
      acc2 = __builtin_amdgcn_wmma_f32_16x16x32_bf16(false, a.v, false, bfrag,
                                                     (short)0, acc2, false, false);
      ar += 16 * LDSROW;
      a.h[0] = *(const v8bf*)ar; a.h[1] = *(const v8bf*)(ar + 16);
      acc3 = __builtin_amdgcn_wmma_f32_16x16x32_bf16(false, a.v, false, bfrag,
                                                     (short)0, acc3, false, false);
    }

    // score contribution: nv[d] * tanh(k + bk + q), reduced over d (16 lanes)
    const float qv  = qbuf[(size_t)b * UD + u];
    const float bkv = bk[u];
    auto emit = [&](v8f acc, int mt) {
#pragma unroll
      for (int r = 0; r < 8; ++r) {
        float e = nvd * tanhf(acc[r] + bkv + qv);
        e += __shfl_xor(e, 1, 16);
        e += __shfl_xor(e, 2, 16);
        e += __shfl_xor(e, 4, 16);
        e += __shfl_xor(e, 8, 16);
        if (lr == 0)
          atomicAdd(&sAdd[(mt * 16 + half * 8 + r) * HH + h], e);
      }
    };
    emit(acc0, 0); emit(acc1, 1); emit(acc2, 2); emit(acc3, 3);
  }
  __syncthreads();

  // ---- softmax over heads, per t-row ----
  for (int i = tid; i < TM * HH; i += 256) {
    int tl = i >> 3, h = i & 7;
    const float* ar = &sAdd[tl * HH];
    float mx = ar[0];
#pragma unroll
    for (int j = 1; j < HH; ++j) mx = fmaxf(mx, ar[j]);
    float sum = 0.f;
#pragma unroll
    for (int j = 0; j < HH; ++j) sum += __expf(ar[j] - mx);
    sW[i] = __expf(ar[h] - mx) / sum;
  }
  __syncthreads();

  // ---- context: ctx[b,h,d] += sum_t w[t,h] * key[b,t,d] (fp32 key, cached) ----
  for (int o = tid; o < HH * KD; o += 256) {
    int h = o >> 9;                    // KD == 512
    int d = o & (KD - 1);
    float s = 0.f;
#pragma unroll 4
    for (int tl = 0; tl < TM; ++tl)
      s += sW[tl * HH + h] * keyTile[(size_t)tl * KD + d];
    atomicAdd(&out[((size_t)b * HH + h) * KD + d], s);
  }
}

// --------------------------- launcher --------------------------------------

extern "C" void kernel_launch(void* const* d_in, const int* in_sizes, int n_in,
                              void* d_out, int out_size, void* d_ws, size_t ws_size,
                              hipStream_t stream) {
  (void)in_sizes; (void)n_in; (void)out_size; (void)ws_size;
  const float* query = (const float*)d_in[0];   // [64,1,512]
  const float* key   = (const float*)d_in[1];   // [64,2048,512]
  const float* Wq    = (const float*)d_in[2];   // [1024,512]
  const float* bq    = (const float*)d_in[3];   // [1024]
  const float* Wk    = (const float*)d_in[4];   // [1024,512]
  const float* bk    = (const float*)d_in[5];   // [1024]
  const float* v     = (const float*)d_in[6];   // [128]
  float* out = (float*)d_out;                   // [64, 4096]

  // workspace layout: Wk bf16 (1 MiB) | q fp32 (256 KiB) | nv (512 B)
  __bf16* wkb  = (__bf16*)d_ws;
  float*  qbuf = (float*)((char*)d_ws + (1u << 20));
  float*  nvb  = (float*)((char*)d_ws + (1u << 20) + 262144u);

  cvt_wk_bf16_kernel<<<512, 256, 0, stream>>>(Wk, wkb);        // 1024*512 elems
  normv_kernel<<<1, DH, 0, stream>>>(v, nvb);
  qproj_kernel<<<NB, 256, 0, stream>>>(query, Wq, bq, qbuf);
  zero_out_kernel<<<256, 256, 0, stream>>>(out);               // 64*4096 floats
  attn_fused_kernel<<<dim3(TK / TM, NB), 256, 0, stream>>>(key, wkb, qbuf, nvb, bk, out);
}
